// AFTEncoderBlock_37830071943853
// MI455X (gfx1250) — matrix operand order repaired and checked by reading, not compiled, not run on hardware
//
#include <hip/hip_runtime.h>
#include <hip/hip_bf16.h>
#include <math.h>
#include <stdint.h>

typedef __attribute__((ext_vector_type(16))) _Float16 v16h;
typedef __attribute__((ext_vector_type(8)))  _Float16 v8h;
typedef __attribute__((ext_vector_type(8)))  float    v8f;
typedef __attribute__((ext_vector_type(4)))  float    v4f;

#define BB 8
#define TT 4096
#define DDIM 512
#define HDIM 1024
#define MROWS (BB*TT)      /* 32768 */
#define THH (TT*HDIM)      /* 4194304 = 2^22 */

enum { EP_SIGMOID=0, EP_EXP=1, EP_BIAS=2, EP_RESID=3, EP_GELU=4, EP_SCALE2=5 };

#if __has_builtin(__builtin_amdgcn_global_load_async_to_lds_b32)
#define HAS_ASYNC_LDS 1
typedef __attribute__((address_space(1))) int gas_int;   // global addrspace
typedef __attribute__((address_space(3))) int las_int;   // LDS addrspace
#else
#define HAS_ASYNC_LDS 0
#endif

// ---------------------------------------------------------------------------
// Weight prep: f32 [R,C] row-major  ->  f16 [C,R] row-major (N-major for GEMM B)
// Stages the input tile into LDS via the CDNA5 async global->LDS path when
// available (ASYNCcnt-tracked), else plain stores.
// ---------------------------------------------------------------------------
__global__ __launch_bounds__(256)
void transpose_f32_f16(const float* __restrict__ in, _Float16* __restrict__ out,
                       int R, int C) {
    __shared__ float tile[32][33];
    const int bc = blockIdx.x * 32;          // col base in input
    const int br = blockIdx.y * 32;          // row base in input
    const int tx = threadIdx.x & 31;
    const int ty = threadIdx.x >> 5;         // 0..7
#if HAS_ASYNC_LDS
    #pragma unroll
    for (int i = 0; i < 32; i += 8) {
        const float* gsrc = in + (size_t)(br + ty + i) * C + (bc + tx);
        __builtin_amdgcn_global_load_async_to_lds_b32(
            (gas_int*)(uintptr_t)gsrc,
            (las_int*)(uintptr_t)&tile[ty + i][tx],
            0, 0);
    }
    asm volatile("s_wait_asynccnt 0" ::: "memory");
#else
    #pragma unroll
    for (int i = 0; i < 32; i += 8)
        tile[ty + i][tx] = in[(size_t)(br + ty + i) * C + (bc + tx)];
#endif
    __syncthreads();
    #pragma unroll
    for (int i = 0; i < 32; i += 8)
        out[(size_t)(bc + ty + i) * R + (br + tx)] = (_Float16)tile[tx][ty + i];
}

// ---------------------------------------------------------------------------
// LayerNorm: one wave32 per row of 512; 16 elems/lane; shfl_xor reduction.
// ---------------------------------------------------------------------------
__global__ __launch_bounds__(256)
void ln_kernel(const float* __restrict__ x, const float* __restrict__ g,
               const float* __restrict__ b, float* __restrict__ yf,
               _Float16* __restrict__ yh, int rows) {
    const int wave = (blockIdx.x * blockDim.x + threadIdx.x) >> 5;
    const int lane = threadIdx.x & 31;
    if (wave >= rows) return;
    const float* xr = x + (size_t)wave * DDIM;
    v4f v[4];
    float s = 0.f, s2 = 0.f;
    #pragma unroll
    for (int i = 0; i < 4; ++i) {
        v[i] = *(const v4f*)(xr + lane * 16 + i * 4);
        #pragma unroll
        for (int j = 0; j < 4; ++j) { s += v[i][j]; s2 += v[i][j] * v[i][j]; }
    }
    #pragma unroll
    for (int off = 16; off > 0; off >>= 1) {
        s  += __shfl_xor(s,  off, 32);
        s2 += __shfl_xor(s2, off, 32);
    }
    const float mu  = s * (1.f / DDIM);
    const float var = s2 * (1.f / DDIM) - mu * mu;
    const float rs  = rsqrtf(var + 1e-5f);
    #pragma unroll
    for (int i = 0; i < 4; ++i) {
        const int base = lane * 16 + i * 4;
        const v4f gg = *(const v4f*)(g + base);
        const v4f bb = *(const v4f*)(b + base);
        #pragma unroll
        for (int j = 0; j < 4; ++j) {
            const float y = (v[i][j] - mu) * rs * gg[j] + bb[j];
            if (yf) yf[(size_t)wave * DDIM + base + j] = y;
            yh[(size_t)wave * DDIM + base + j] = (_Float16)y;
        }
    }
}

// ---------------------------------------------------------------------------
// WMMA GEMM: C[M,N] = A[M,K](f16,row-major) x Bt[N,K]^T(f16,N-major) + epilogue
// Block = 256 threads = 8 waves (2x4): block tile 64(M) x 256(N).
// Each wave computes a 32x64 tile = 2x4 v_wmma_f32_16x16x32_f16 frags
// (8 WMMA per 12 b128 fragment loads per K-step).
// ---------------------------------------------------------------------------
template<int MODE>
__global__ __launch_bounds__(256)
void gemm_wmma(const _Float16* __restrict__ A, const _Float16* __restrict__ Bt,
               const float* __restrict__ bias, const float* __restrict__ extra,
               float* __restrict__ outF, _Float16* __restrict__ outH,
               int Mdim, int N, int K) {
    const int lane = threadIdx.x & 31;
    const int wave = threadIdx.x >> 5;
    const int m0 = blockIdx.x * 64  + (wave >> 2) * 32;   // wave M origin
    const int n0 = blockIdx.y * 256 + (wave & 3) * 64;    // wave N origin
    const int lm = lane & 15;
    const int lg = lane >> 4;

    v8f acc[2][4] = {};

    const _Float16* Ar[2];
    Ar[0] = A + (size_t)(m0 + lm)      * K;
    Ar[1] = A + (size_t)(m0 + 16 + lm) * K;
    const _Float16* Br[4];
    #pragma unroll
    for (int j = 0; j < 4; ++j) Br[j] = Bt + (size_t)(n0 + 16 * j + lm) * K;

    union Frag { v16h v; v8h h[2]; };

    for (int k0 = 0; k0 < K; k0 += 32) {
        const int ka = k0 + lg * 8;    // A frag: K 0-7/16-23 (lanes 0-15), 8-15/24-31 (lanes 16-31)
        const int kb = k0 + lg * 8;    // B frag: K 0-7/16-23 (lanes 0-15), 8-15/24-31 (lanes 16-31)
        Frag a[2], b[4];
        #pragma unroll
        for (int i = 0; i < 2; ++i) {
            a[i].h[0] = *(const v8h*)(Ar[i] + ka);
            a[i].h[1] = *(const v8h*)(Ar[i] + ka + 16);
        }
        #pragma unroll
        for (int j = 0; j < 4; ++j) {
            b[j].h[0] = *(const v8h*)(Br[j] + kb);
            b[j].h[1] = *(const v8h*)(Br[j] + kb + 16);
        }
        if (k0 + 32 < K) {   // keep the A stream warm (global_prefetch_b8)
            __builtin_prefetch(Ar[0] + ka + 32, 0, 3);
            __builtin_prefetch(Ar[1] + ka + 32, 0, 3);
        }
        #pragma unroll
        for (int i = 0; i < 2; ++i)
            #pragma unroll
            for (int j = 0; j < 4; ++j)
                acc[i][j] = __builtin_amdgcn_wmma_f32_16x16x32_f16(
                    false, a[i].v, false, b[j].v, (short)0, acc[i][j], false, false);
    }

    #pragma unroll
    for (int i = 0; i < 2; ++i) {
        #pragma unroll
        for (int j = 0; j < 4; ++j) {
            const int col = n0 + 16 * j + lm;
            const float bs = bias[col];
            float wb = 0.f;
            if constexpr (MODE == EP_EXP) wb = extra[col];
            #pragma unroll
            for (int r = 0; r < 8; ++r) {
                const int row = m0 + 16 * i + lg * 8 + r;  // C layout: m = r + 8*(lane>>4)
                const size_t idx = (size_t)row * N + col;
                const float v = acc[i][j][r] + bs;
                if constexpr (MODE == EP_SIGMOID) {
                    outH[idx] = (_Float16)(1.f / (1.f + __expf(-v)));
                } else if constexpr (MODE == EP_EXP) {
                    outF[idx] = __expf(v + wb);
                } else if constexpr (MODE == EP_BIAS) {
                    outF[idx] = v;
                } else if constexpr (MODE == EP_RESID) {
                    outF[idx] = v + extra[idx];
                } else if constexpr (MODE == EP_GELU) {
                    outH[idx] = (_Float16)(0.5f * v * (1.f + erff(v * 0.70710678118f)));
                } else { // EP_SCALE2
                    outF[idx] = 2.f * v;
                }
            }
        }
    }
}

// ---------------------------------------------------------------------------
// AFT batch reduction: weighted[t,h] = sum_b numer*V / sum_b numer
// ---------------------------------------------------------------------------
__global__ __launch_bounds__(256)
void aft_reduce(const float* __restrict__ numer, const float* __restrict__ vmat,
                float* __restrict__ weighted, int th) {
    const int i = blockIdx.x * blockDim.x + threadIdx.x;
    if (i >= th) return;
    float dn = 0.f, wn = 0.f;
    #pragma unroll
    for (int b = 0; b < BB; ++b) {
        const size_t o = (size_t)b * th + i;
        const float nb = numer[o];
        dn += nb;
        wn += nb * vmat[o];
    }
    weighted[i] = wn / dn;
}

// Yt[b,t,h] = sigmoid(Q) * weighted[t,h]  (broadcast over b), as f16 GEMM input
__global__ __launch_bounds__(256)
void yt_kernel(const _Float16* __restrict__ qsig, const float* __restrict__ weighted,
               _Float16* __restrict__ yt, int total) {
    const int i = blockIdx.x * blockDim.x + threadIdx.x;
    if (i >= total) return;
    yt[i] = (_Float16)((float)qsig[i] * weighted[i & (THH - 1)]);
}

// ---------------------------------------------------------------------------
extern "C" void kernel_launch(void* const* d_in, const int* in_sizes, int n_in,
                              void* d_out, int out_size, void* d_ws, size_t ws_size,
                              hipStream_t stream) {
    (void)in_sizes; (void)n_in; (void)out_size; (void)ws_size;
    const float* x     = (const float*)d_in[0];
    const float* gamma = (const float*)d_in[1];
    const float* beta  = (const float*)d_in[2];
    const float* Wq    = (const float*)d_in[3];
    const float* bq    = (const float*)d_in[4];
    const float* Wk    = (const float*)d_in[5];
    const float* bk    = (const float*)d_in[6];
    const float* Wv    = (const float*)d_in[7];
    const float* bv    = (const float*)d_in[8];
    const float* wbias = (const float*)d_in[9];
    const float* Wo    = (const float*)d_in[10];
    const float* bo    = (const float*)d_in[11];
    const float* W1    = (const float*)d_in[12];
    const float* b1    = (const float*)d_in[13];
    const float* W2    = (const float*)d_in[14];
    const float* b2    = (const float*)d_in[15];
    float* out = (float*)d_out;

    char* ws = (char*)d_ws;
    const size_t MB = 1024ull * 1024ull;
    float*    x1f   = (float*)   (ws + 0);        //  64 MB  LN(x) f32 (residual)
    _Float16* x1h   = (_Float16*)(ws + 64*MB);    //  32 MB  LN(x) f16 (GEMM A)
    _Float16* wqt   = (_Float16*)(ws + 96*MB);    //   1 MB each, f16 N-major
    _Float16* wkt   = (_Float16*)(ws + 97*MB);
    _Float16* wvt   = (_Float16*)(ws + 98*MB);
    _Float16* wot   = (_Float16*)(ws + 99*MB);
    _Float16* w1t   = (_Float16*)(ws + 100*MB);
    _Float16* w2t   = (_Float16*)(ws + 101*MB);
    _Float16* qsig  = (_Float16*)(ws + 102*MB);   //  64 MB
    float*    numer = (float*)   (ws + 166*MB);   // 128 MB
    float*    vbuf  = (float*)   (ws + 294*MB);   // 128 MB
    float*    wgt   = (float*)   (ws + 422*MB);   //  16 MB  weighted [T,H]
    _Float16* ytb   = (_Float16*)(ws + 166*MB);   //  64 MB  (reuses numer)
    float*    x2    = (float*)   (ws + 294*MB);   //  64 MB  (reuses vbuf)
    _Float16* x3h   = (_Float16*)(ws + 358*MB);   //  32 MB  (reuses vbuf tail)
    _Float16* hbuf  = (_Float16*)(ws + 102*MB);   //  64 MB  (reuses qsig)

    dim3 blk(256);

    // --- weight prep (f32 -> f16, transposed to N-major) ---
    transpose_f32_f16<<<dim3(HDIM/32, DDIM/32), blk, 0, stream>>>(Wq, wqt, DDIM, HDIM);
    transpose_f32_f16<<<dim3(HDIM/32, DDIM/32), blk, 0, stream>>>(Wk, wkt, DDIM, HDIM);
    transpose_f32_f16<<<dim3(HDIM/32, DDIM/32), blk, 0, stream>>>(Wv, wvt, DDIM, HDIM);
    transpose_f32_f16<<<dim3(DDIM/32, HDIM/32), blk, 0, stream>>>(Wo, wot, HDIM, DDIM);
    transpose_f32_f16<<<dim3(HDIM/32, DDIM/32), blk, 0, stream>>>(W1, w1t, DDIM, HDIM);
    transpose_f32_f16<<<dim3(DDIM/32, HDIM/32), blk, 0, stream>>>(W2, w2t, HDIM, DDIM);

    // --- pre-LN ---
    ln_kernel<<<MROWS/8, blk, 0, stream>>>(x, gamma, beta, x1f, x1h, MROWS);

    // --- QKV GEMMs with fused epilogues ---
    gemm_wmma<EP_SIGMOID><<<dim3(MROWS/64, HDIM/256), blk, 0, stream>>>(
        x1h, wqt, bq, nullptr, nullptr, qsig, MROWS, HDIM, DDIM);
    gemm_wmma<EP_EXP><<<dim3(MROWS/64, HDIM/256), blk, 0, stream>>>(
        x1h, wkt, bk, wbias, numer, nullptr, MROWS, HDIM, DDIM);
    gemm_wmma<EP_BIAS><<<dim3(MROWS/64, HDIM/256), blk, 0, stream>>>(
        x1h, wvt, bv, nullptr, vbuf, nullptr, MROWS, HDIM, DDIM);

    // --- AFT batch reduction + broadcast multiply ---
    aft_reduce<<<THH/256, blk, 0, stream>>>(numer, vbuf, wgt, THH);
    yt_kernel<<<(MROWS*HDIM)/256, blk, 0, stream>>>(qsig, wgt, ytb, MROWS*HDIM);

    // --- output projection + residual ---
    gemm_wmma<EP_RESID><<<dim3(MROWS/64, DDIM/256), blk, 0, stream>>>(
        ytb, wot, bo, x1f, x2, nullptr, MROWS, DDIM, HDIM);

    // --- second LN + MLP ---
    ln_kernel<<<MROWS/8, blk, 0, stream>>>(x2, gamma, beta, nullptr, x3h, MROWS);
    gemm_wmma<EP_GELU><<<dim3(MROWS/64, HDIM/256), blk, 0, stream>>>(
        x3h, w1t, b1, nullptr, nullptr, hbuf, MROWS, HDIM, DDIM);
    gemm_wmma<EP_SCALE2><<<dim3(MROWS/64, DDIM/256), blk, 0, stream>>>(
        hbuf, w2t, b2, nullptr, out, nullptr, MROWS, DDIM, HDIM);
}
